// CapsuleLayer_70789650973135
// MI455X (gfx1250) — compile-verified, matrix operand order checked by reading, not run on hardware
//
#include <hip/hip_runtime.h>
#include <hip/hip_bf16.h>
#include <stdint.h>

// CapsuleLayer routing for MI455X (gfx1250), wave32 + WMMA bf16 + async LDS.
// Shapes: x[B=64][N=1152][I=64], W[C=32][N=1152][O=64][I=64]
// u_hat layout [N][C][B][O]: each (c,n) GEMM tile is a contiguous 16KB block
// and routing loads are coalesced over O.
// Workspace requirement: 604MB (u_hat) + 2 * 512KB.

typedef __attribute__((ext_vector_type(16))) __bf16 v16bf;
typedef __attribute__((ext_vector_type(8)))  float  v8f;

#define B_DIM 64
#define C_DIM 32
#define N_DIM 1152
#define O_DIM 64
#define I_DIM 64
#define NCHUNK_G 9     // n per GEMM block (1152 = 128 * 9)
#define NCHUNK_R 36    // n per routing block (1152 = 32 * 36)

__device__ __forceinline__ __bf16 f2bf(float f) {
  // native fp32 -> bf16 (RNE): lets clang emit v_cvt(_pk)_bf16_f32
  return (__bf16)f;
}

// Async copy of one 64x64 fp32 W tile (16KB) global -> LDS (ASYNCcnt-tracked).
// 128 threads x 8 x b128 = 4096 floats. Row stride in LDS = 68 floats (16B-
// aligned pad). Completion: s_wait_asynccnt (in-order for async loads).
__device__ __forceinline__ void async_copy_w_tile(const float* __restrict__ wb,
                                                  float (*dst)[68], int tid) {
  #pragma unroll
  for (int j = 0; j < 8; ++j) {
    const int q   = tid + j * 128;      // 0..1023 16B chunks
    const int row = q >> 4;
    const int c4  = (q & 15) << 2;
    const unsigned     lds = (unsigned)(uintptr_t)&dst[row][c4];
    const float* g = wb + row * I_DIM + c4;
    asm volatile("global_load_async_to_lds_b128 %0, %1, off"
                 :: "v"(lds), "v"(g) : "memory");
  }
}

// ---------------------------------------------------------------------------
// u_hat GEMM: per (c, n): u[b][o] = sum_i x[b][n][i] * W[c][n][o][i]
// fused epilogue: s0[b][c][o] += sum_n u[b][o]  (atomic partial flush)
// W tiles double-buffered via async global->LDS, overlapped with WMMA.
// ---------------------------------------------------------------------------
__global__ __launch_bounds__(128) void capsule_uhat_gemm(
    const float* __restrict__ x,      // [B, N, I]
    const float* __restrict__ w,      // [C, N, O, I]
    float*       __restrict__ uhat,   // [N, C, B, O]
    float*       __restrict__ s0)     // [B, C, O] (pre-zeroed, atomic accum)
{
  __shared__ __bf16 Ash[64][72];      // x tile  [b][i], bf16 (pad: no bank cf)
  __shared__ float  Wsh[2][64][68];   // W tiles [o][i], fp32, double-buffered

  const int c    = blockIdx.x;
  const int n0   = blockIdx.y * NCHUNK_G;
  const int tid  = threadIdx.x;
  const int wave = tid >> 5;
  const int lane = tid & 31;
  const int mb   = wave * 16;          // b-tile base for this wave
  const int mrow = mb + (lane & 15);   // A-matrix row (ISA 16-bit A layout)
  const int hl   = lane >> 4;          // lane half selects K sub-block
  const int ncol = lane & 15;          // C/D column (O index)

  const float* wbase = w + ((size_t)c * N_DIM + n0) * (O_DIM * I_DIM);

  v8f s0acc[4];
  #pragma unroll
  for (int ot = 0; ot < 4; ++ot) s0acc[ot] = v8f{};

  // prologue: start fetching W tile for n0
  async_copy_w_tile(wbase, Wsh[0], tid);

  for (int nn = 0; nn < NCHUNK_G; ++nn) {
    const int n = n0 + nn;

    // kick off next W tile into the other buffer (overlaps with compute)
    if (nn + 1 < NCHUNK_G)
      async_copy_w_tile(wbase + (size_t)(nn + 1) * (O_DIM * I_DIM),
                        Wsh[(nn + 1) & 1], tid);

    // cooperative x-tile load (L2-warm, shared by all 32 c-blocks), fp32->bf16
    const float* xb = x + (size_t)n * I_DIM;                     // + b*(N*I)
    for (int e = tid * 4; e < 64 * 64; e += 128 * 4) {
      const int row = e >> 6, col = e & 63;
      const float4 xv = *(const float4*)(xb + (size_t)row * (N_DIM * I_DIM) + col);
      Ash[row][col + 0] = f2bf(xv.x);
      Ash[row][col + 1] = f2bf(xv.y);
      Ash[row][col + 2] = f2bf(xv.z);
      Ash[row][col + 3] = f2bf(xv.w);
    }

    // wait for the CURRENT W tile: async loads complete in order, so with the
    // next tile's 8 ops in flight, asynccnt<=8 means this tile has landed.
    if (nn + 1 < NCHUNK_G)
      asm volatile("s_wait_asynccnt 0x8" ::: "memory");
    else
      asm volatile("s_wait_asynccnt 0x0" ::: "memory");
    __syncthreads();

    const float(*Wt)[68] = Wsh[nn & 1];

    // A fragments (shared across the 4 o-tiles), K-split per ISA layout:
    // lanes 0-15: K = e&7 (+16 for e>=8); lanes 16-31: +8
    v16bf a0, a1;
    #pragma unroll
    for (int e = 0; e < 16; ++e) {
      const int k = (e & 7) + ((e & 8) ? 16 : 0) + hl * 8;
      a0[e] = Ash[mrow][k];
      a1[e] = Ash[mrow][k + 32];
    }

    v8f acc[4];
    #pragma unroll
    for (int ot = 0; ot < 4; ++ot) {
      const int ob = ot * 16 + ncol;
      v16bf b0, b1;
      #pragma unroll
      for (int e = 0; e < 16; ++e) {
        const int k = (e & 7) + ((e & 8) ? 16 : 0) + hl * 8;
        b0[e] = f2bf(Wt[ob][k]);        // B[k][o] = W[o][i=k], cvt at gather
        b1[e] = f2bf(Wt[ob][k + 32]);
      }
      v8f cacc = v8f{};
      cacc = __builtin_amdgcn_wmma_f32_16x16x32_bf16(
          false, a0, false, b0, (short)0, cacc, false, false);
      cacc = __builtin_amdgcn_wmma_f32_16x16x32_bf16(
          false, a1, false, b1, (short)0, cacc, false, false);
      acc[ot] = cacc;
      s0acc[ot] += cacc;
    }

    // store u_hat tile: C/D layout row m = hl*8 + r
    float* ub = uhat + ((size_t)n * C_DIM + c) * (B_DIM * O_DIM);
    #pragma unroll
    for (int ot = 0; ot < 4; ++ot) {
      #pragma unroll
      for (int r = 0; r < 8; ++r) {
        const int brow = mb + hl * 8 + r;
        ub[brow * O_DIM + ot * 16 + ncol] = acc[ot][r];
      }
    }
    __syncthreads();
  }

  // flush Σ_n partials into s0[b][c][o]
  #pragma unroll
  for (int ot = 0; ot < 4; ++ot) {
    #pragma unroll
    for (int r = 0; r < 8; ++r) {
      const int brow = mb + hl * 8 + r;
      atomicAdd(&s0[((size_t)brow * C_DIM + c) * O_DIM + ot * 16 + ncol],
                s0acc[ot][r]);
    }
  }
}

// ---------------------------------------------------------------------------
// Routing pass: s[b][c][o] += Σ_n softmax_c(u_hat * coeff) * u_hat
// thread owns (b, o) and all 32 c-values in registers.
// ---------------------------------------------------------------------------
__global__ __launch_bounds__(256) void capsule_route(
    const float* __restrict__ uhat,   // [N, C, B, O]
    const float* __restrict__ coeff,  // [B, C, O]  (v0 or v0+v1)
    float*       __restrict__ sout)   // [B, C, O]  (pre-zeroed, atomic accum)
{
  const int tid = threadIdx.x;
  const int b   = blockIdx.x * 4 + (tid >> 6);
  const int o   = tid & 63;
  const int n0  = blockIdx.y * NCHUNK_R;

  float v[C_DIM], acc[C_DIM];
  #pragma unroll
  for (int c = 0; c < C_DIM; ++c) {
    v[c]   = coeff[((size_t)b * C_DIM + c) * O_DIM + o];
    acc[c] = 0.f;
  }

  for (int nn = 0; nn < NCHUNK_R; ++nn) {
    const int n = n0 + nn;
    const float* ub = uhat + (size_t)n * (C_DIM * B_DIM * O_DIM)
                           + (size_t)b * O_DIM + o;
    float u[C_DIM], e[C_DIM];
    float m = -3.4e38f;
    #pragma unroll
    for (int c = 0; c < C_DIM; ++c) {
      u[c] = ub[(size_t)c * (B_DIM * O_DIM)];
      e[c] = u[c] * v[c];
      m = fmaxf(m, e[c]);
    }
    float sum = 0.f;
    #pragma unroll
    for (int c = 0; c < C_DIM; ++c) { e[c] = __expf(e[c] - m); sum += e[c]; }
    const float inv = 1.f / sum;
    #pragma unroll
    for (int c = 0; c < C_DIM; ++c) acc[c] += u[c] * (e[c] * inv);
  }

  #pragma unroll
  for (int c = 0; c < C_DIM; ++c)
    atomicAdd(&sout[((size_t)b * C_DIM + c) * O_DIM + o], acc[c]);
}

// ---------------------------------------------------------------------------
// squash: v = (s²/(1+s²)) * s / sqrt(s² + eps), s pre-scaled
// mode 0: coeff = v     (v0)
// mode 1: coeff += v    (v0 + v1)
// mode 2: out = v       (final v2)
// ---------------------------------------------------------------------------
__global__ void capsule_squash(const float* __restrict__ s, float scale,
                               float* __restrict__ coeff,
                               float* __restrict__ out, int mode, int n)
{
  const int i = blockIdx.x * blockDim.x + threadIdx.x;
  if (i >= n) return;
  const float sv = s[i] * scale;
  const float sq = sv * sv;
  const float vv = (sq / (1.f + sq)) * sv * rsqrtf(sq + 1e-8f);
  if (mode == 0)      coeff[i] = vv;
  else if (mode == 1) coeff[i] += vv;
  else                out[i] = vv;
}

// ---------------------------------------------------------------------------
extern "C" void kernel_launch(void* const* d_in, const int* in_sizes, int n_in,
                              void* d_out, int out_size, void* d_ws, size_t ws_size,
                              hipStream_t stream) {
  (void)in_sizes; (void)n_in; (void)out_size; (void)ws_size;
  const float* x = (const float*)d_in[0];  // [64, 1152, 64]
  const float* w = (const float*)d_in[1];  // [32, 1152, 64, 64]
  float* out = (float*)d_out;              // [64, 32, 64]

  const size_t UHAT_ELEMS = (size_t)N_DIM * C_DIM * B_DIM * O_DIM; // 150,994,944
  const int    SN         = B_DIM * C_DIM * O_DIM;                 // 131,072

  char*  ws    = (char*)d_ws;
  float* uhat  = (float*)ws;
  float* sbuf  = (float*)(ws + UHAT_ELEMS * sizeof(float));
  float* coeff = sbuf + SN;

  const dim3 gGemm(C_DIM, N_DIM / NCHUNK_G);      // 32 x 128 blocks, 128 thr
  const dim3 gRoute(B_DIM / 4, N_DIM / NCHUNK_R); // 16 x 32 blocks, 256 thr
  const int  gSq = (SN + 255) / 256;

  // iter 0: u_hat + Σ_n (fused); c_ij = 1/32 -> scale s0 by 1/32; v0 -> coeff
  hipMemsetAsync(sbuf, 0, SN * sizeof(float), stream);
  capsule_uhat_gemm<<<gGemm, 128, 0, stream>>>(x, w, uhat, sbuf);
  capsule_squash<<<gSq, 256, 0, stream>>>(sbuf, 1.f / 32.f, coeff, out, 0, SN);

  // iter 1: b1 = u_hat*v0 -> softmax -> s1; v1; coeff = v0+v1
  hipMemsetAsync(sbuf, 0, SN * sizeof(float), stream);
  capsule_route<<<gRoute, 256, 0, stream>>>(uhat, coeff, sbuf);
  capsule_squash<<<gSq, 256, 0, stream>>>(sbuf, 1.f, coeff, out, 1, SN);

  // iter 2: b2 = u_hat*(v0+v1) -> softmax -> s2; out = squash(s2)
  hipMemsetAsync(sbuf, 0, SN * sizeof(float), stream);
  capsule_route<<<gRoute, 256, 0, stream>>>(uhat, coeff, sbuf);
  capsule_squash<<<gSq, 256, 0, stream>>>(sbuf, 1.f, coeff, out, 2, SN);
}
